// ContrastiveLoss_21449066676923
// MI455X (gfx1250) — matrix operand order repaired and checked by reading, not compiled, not run on hardware
//
#include <hip/hip_runtime.h>

#define NN 8192
#define DD 128

typedef _Float16 h4   __attribute__((ext_vector_type(4)));
typedef _Float16 h8   __attribute__((ext_vector_type(8)));
typedef _Float16 v16h __attribute__((ext_vector_type(16)));
typedef float    v8f  __attribute__((ext_vector_type(8)));

__device__ __forceinline__ v16h pack16(h8 lo, h8 hi) {
  v16h r;
#pragma unroll
  for (int i = 0; i < 8; ++i) { r[i] = lo[i]; r[i + 8] = hi[i]; }
  return r;
}

// ---------------------------------------------------------------------------
// Kernel 0: row-normalize x (f32 [8192][128]) -> xh (f16 [8192][128])
// One wave per row; lane l holds cols 4l..4l+3.
// ---------------------------------------------------------------------------
__global__ __launch_bounds__(256) void normalize_rows(
    const float* __restrict__ x, _Float16* __restrict__ xh) {
  const int row = blockIdx.x * 8 + (threadIdx.x >> 5);
  const int l = threadIdx.x & 31;
  const float4 v = ((const float4*)(x + (size_t)row * DD))[l];
  float s = v.x * v.x + v.y * v.y + v.z * v.z + v.w * v.w;
#pragma unroll
  for (int m = 16; m >= 1; m >>= 1) s += __shfl_xor(s, m, 32);
  const float r = __builtin_amdgcn_rsqf(s);
  h4 o;
  o[0] = (_Float16)(v.x * r);
  o[1] = (_Float16)(v.y * r);
  o[2] = (_Float16)(v.z * r);
  o[3] = (_Float16)(v.w * r);
  ((h4*)(xh + (size_t)row * DD))[l] = o;
}

// ---------------------------------------------------------------------------
// Main tiled pass. Block = 256 thr = 8 waves, owns 16 output rows.
// Wave w sweeps column tiles j0 = 16w, 16w+128, ... (64 iterations).
// PASS 1: per-row S (full exp sum), P (positive exp sum), T (sum y*t, log2
//         units), Np (positive count).
// PASS 2: per-row U = sum y*log2(e + nl), with nl = S - P.
// y loads are non-temporal (268 MB streamed, zero reuse); xh stays in L2.
// Cross-wave merge is a fixed-order LDS reduction => bit-deterministic.
// ---------------------------------------------------------------------------
template <int PASS>
__global__ __launch_bounds__(256) void pass_kernel(
    const _Float16* __restrict__ xh, const float* __restrict__ y,
    float* __restrict__ S, float* __restrict__ P, float* __restrict__ Tt,
    float* __restrict__ Np, float* __restrict__ U) {
  __shared__ float sS[8][16], sP[8][16], sT[8][16], sN[8][16], sU[8][16];
  const int tid = threadIdx.x;
  const int lane = tid & 31;
  const int li = lane & 15;   // = N column within tile; = M row for A loads
  const int half = lane >> 4;
  const int wave = tid >> 5;
  const int r0 = blockIdx.x << 4;

  // A tile: rows r0..r0+15, K=0..127 (4 chunks of 32).
  // 16-bit A 16x32 layout: lanes 0-15 hold K{0-7,16-23}, lanes 16-31 K{8-15,24-31}.
  v16h A[4];
  {
    const _Float16* ap = xh + (size_t)(r0 + li) * DD + half * 8;
#pragma unroll
    for (int kt = 0; kt < 4; ++kt)
      A[kt] = pack16(*(const h8*)(ap + kt * 32), *(const h8*)(ap + kt * 32 + 16));
  }

  const float* yrow[8];
  float nlrow[8];
#pragma unroll
  for (int v = 0; v < 8; ++v) {
    const int r = r0 + v + half * 8;  // C/D layout: VGPR v -> row v + 8*half
    yrow[v] = y + (size_t)r * NN;
    if (PASS == 2) nlrow[v] = S[r] - P[r];
  }

  float a0[8], a1[8], a2[8], a3[8];
#pragma unroll
  for (int v = 0; v < 8; ++v) { a0[v] = 0.f; a1[v] = 0.f; a2[v] = 0.f; a3[v] = 0.f; }

  for (int j0 = wave * 16; j0 < NN; j0 += 16 * 8) {
    // B tile: cols j0..j0+15 (rows of xh, since B = x_norm^T).
    // 16-bit B 32x16 layout: lanes 0-15 hold K{0-7,16-23}, lanes 16-31 K{8-15,24-31}.
    const _Float16* bp = xh + (size_t)(j0 + li) * DD + half * 8;
    v8f c = {};
#pragma unroll
    for (int kt = 0; kt < 4; ++kt) {
      v16h b = pack16(*(const h8*)(bp + kt * 32), *(const h8*)(bp + kt * 32 + 16));
      c = __builtin_amdgcn_wmma_f32_16x16x32_f16(false, A[kt], false, b,
                                                 (short)0, c, false, false);
    }
#pragma unroll
    for (int v = 0; v < 8; ++v) {
      const float t = c[v] * 28.853900817779268f;  // (1/temp)/ln2 = 20*log2(e)
      const float e = __builtin_amdgcn_exp2f(t);   // v_exp_f32
      const float yv = __builtin_nontemporal_load(yrow[v] + j0 + li);
      if (PASS == 1) {
        a0[v] += e;
        a1[v] += e * yv;
        a2[v] += t * yv;
        a3[v] += yv;
      } else {
        a0[v] += yv * __builtin_amdgcn_logf(e + nlrow[v]);  // v_log_f32 = log2
      }
    }
  }

  // Reduce over the 16 lanes sharing each row (xor masks stay within halves).
#pragma unroll
  for (int m = 1; m < 16; m <<= 1) {
#pragma unroll
    for (int v = 0; v < 8; ++v) {
      a0[v] += __shfl_xor(a0[v], m, 32);
      if (PASS == 1) {
        a1[v] += __shfl_xor(a1[v], m, 32);
        a2[v] += __shfl_xor(a2[v], m, 32);
        a3[v] += __shfl_xor(a3[v], m, 32);
      }
    }
  }
  // Per-wave staging: lanes 0 and 16 each write their half's 8 rows.
  // Every slot s*[wave][0..15] is written exactly once -> no init, no atomics.
  if (li == 0) {
#pragma unroll
    for (int v = 0; v < 8; ++v) {
      const int rl = v + half * 8;
      if (PASS == 1) {
        sS[wave][rl] = a0[v];
        sP[wave][rl] = a1[v];
        sT[wave][rl] = a2[v];
        sN[wave][rl] = a3[v];
      } else {
        sU[wave][rl] = a0[v];
      }
    }
  }
  __syncthreads();
  // Fixed-order cross-wave sum (deterministic across replays).
  if (tid < 16) {
    if (PASS == 1) {
      float t0 = 0.f, t1 = 0.f, t2 = 0.f, t3 = 0.f;
#pragma unroll
      for (int w = 0; w < 8; ++w) {
        t0 += sS[w][tid];
        t1 += sP[w][tid];
        t2 += sT[w][tid];
        t3 += sN[w][tid];
      }
      S[r0 + tid] = t0;
      P[r0 + tid] = t1;
      Tt[r0 + tid] = t2;
      Np[r0 + tid] = t3;
    } else {
      float t0 = 0.f;
#pragma unroll
      for (int w = 0; w < 8; ++w) t0 += sU[w][tid];
      U[r0 + tid] = t0;
    }
  }
}

// ---------------------------------------------------------------------------
// Final reduction: loss = -(ln2/N) * sum_i (T_i - U_i) / Np_i
// ---------------------------------------------------------------------------
__global__ __launch_bounds__(256) void finalize_kernel(
    const float* __restrict__ Tt, const float* __restrict__ U,
    const float* __restrict__ Np, float* __restrict__ out) {
  __shared__ float red[256];
  float s = 0.f;
  for (int i = threadIdx.x; i < NN; i += 256) s += (Tt[i] - U[i]) / Np[i];
  red[threadIdx.x] = s;
  __syncthreads();
  for (int off = 128; off > 0; off >>= 1) {
    if (threadIdx.x < off) red[threadIdx.x] += red[threadIdx.x + off];
    __syncthreads();
  }
  if (threadIdx.x == 0)
    out[0] = -red[0] * 0.6931471805599453f / (float)NN;
}

extern "C" void kernel_launch(void* const* d_in, const int* in_sizes, int n_in,
                              void* d_out, int out_size, void* d_ws, size_t ws_size,
                              hipStream_t stream) {
  (void)in_sizes; (void)n_in; (void)out_size; (void)ws_size;
  const float* x = (const float*)d_in[0];  // [8192, 128] f32
  const float* y = (const float*)d_in[1];  // [8192, 8192] f32
  float* out = (float*)d_out;              // scalar f32

  char* ws = (char*)d_ws;
  _Float16* xh = (_Float16*)ws;                        // 2 MB f16 x_norm
  float* S = (float*)(ws + (size_t)NN * DD * sizeof(_Float16));
  float* P = S + NN;
  float* Tt = P + NN;
  float* Np = Tt + NN;
  float* U = Np + NN;

  normalize_rows<<<NN / 8, 256, 0, stream>>>(x, xh);
  pass_kernel<1><<<NN / 16, 256, 0, stream>>>(xh, y, S, P, Tt, Np, U);
  pass_kernel<2><<<NN / 16, 256, 0, stream>>>(xh, y, S, P, Tt, Np, U);
  finalize_kernel<<<1, 256, 0, stream>>>(Tt, U, Np, out);
}
